// CharBiGRU_82824149336606
// MI455X (gfx1250) — matrix operand order repaired and checked by reading, not compiled, not run on hardware
//
#include <hip/hip_runtime.h>
#include <hip/hip_bf16.h>

// ---------------------------------------------------------------------------
// Bidirectional LSTM, B=64 T=512 D=H=512, fp32 I/O, bf16 WMMA compute.
//   Phase 1: convert x + 4 weight matrices to bf16.
//   Phase 2: pre[d][b][t][4H] = x[b][t] @ Wi_d^T + b_d   (WMMA GEMM, bf16 out)
//   Phase 3: persistent scan kernel: 32 blocks/direction, each owns a 16-wide
//            hidden slice (c state in LDS), exchanges h via ping-pong global
//            bf16 buffers with one agent-scope spin barrier per timestep.
// ---------------------------------------------------------------------------

typedef __bf16 bf16_t;
typedef __attribute__((ext_vector_type(16))) __bf16 v16bf;
typedef __attribute__((ext_vector_type(8)))  __bf16 v8bf;
typedef __attribute__((ext_vector_type(8)))  float  v8f;

#define BB   64      // batch
#define TT   512     // time
#define DD   512     // input dim
#define HH   512     // hidden dim
#define NG   2048    // 4*H gate columns
#define HB   (BB*HH) // one h buffer, elements

__device__ __forceinline__ bf16_t f2bf(float f) {
    unsigned u = __builtin_bit_cast(unsigned, f);
    unsigned r = u + 0x7FFFu + ((u >> 16) & 1u);       // round-to-nearest-even
    unsigned short h = (unsigned short)(r >> 16);
    return __builtin_bit_cast(bf16_t, h);
}
__device__ __forceinline__ float bf2f(bf16_t b) {
    unsigned short h = __builtin_bit_cast(unsigned short, b);
    unsigned u = ((unsigned)h) << 16;
    return __builtin_bit_cast(float, u);
}
__device__ __forceinline__ float sigm(float x) { return 1.0f / (1.0f + __expf(-x)); }

// ---------------------------------------------------------------- init / misc
__global__ void k_init_bar(int* bar) {
    if (threadIdx.x < 64) bar[threadIdx.x] = 0;
}

__global__ void k_row_len(const float* __restrict__ mask, int* __restrict__ len) {
    __shared__ float red[256];
    int b = blockIdx.x;
    float s = 0.f;
    for (int i = threadIdx.x; i < TT; i += 256) s += mask[(size_t)b * TT + i];
    red[threadIdx.x] = s;
    __syncthreads();
    for (int off = 128; off; off >>= 1) {
        if (threadIdx.x < off) red[threadIdx.x] += red[threadIdx.x + off];
        __syncthreads();
    }
    if (threadIdx.x == 0) len[b] = (int)(red[0] + 0.5f);
}

__global__ void k_cvt_bf16(const float* __restrict__ src, bf16_t* __restrict__ dst, int n) {
    for (int i = blockIdx.x * blockDim.x + threadIdx.x; i < n; i += gridDim.x * blockDim.x)
        dst[i] = f2bf(src[i]);
}

// h0 (zeros in setup, but be faithful) into ping buffer 0 of each direction
__global__ void k_init_h(const float* __restrict__ h0, bf16_t* __restrict__ hbuf) {
    int i = blockIdx.x * blockDim.x + threadIdx.x;     // 0 .. 2*HB-1
    int dir = i >> 15;
    int j   = i & (HB - 1);
    hbuf[(size_t)dir * 2 * HB + j] = f2bf(h0[j]);
}

// -------------------------------------------------- Phase 2: input projection
// pre[d][m][n] = sum_k x[m][k] * Wi_d[n][k] + bias_d[n],  m in [0,B*T)
// Block: 256 thr = 8 waves; block tile 64(M) x 32(N); wave tile 16x16; K=512.
__global__ __launch_bounds__(256) void k_pre_gemm(
    const bf16_t* __restrict__ xb,
    const bf16_t* __restrict__ Wif, const bf16_t* __restrict__ Wib,
    const float*  __restrict__ bf_, const float*  __restrict__ bb_,
    bf16_t* __restrict__ pref, bf16_t* __restrict__ preb)
{
    const int dir = blockIdx.z;
    const bf16_t* W    = dir ? Wib  : Wif;
    const float*  bias = dir ? bb_  : bf_;
    bf16_t*       pre  = dir ? preb : pref;

    const int wave = threadIdx.x >> 5, lane = threadIdx.x & 31;
    const int lh = lane & 15, ls = lane >> 4;          // lane-in-16, half select
    const int m0 = blockIdx.x * 64 + (wave & 3) * 16;
    const int n0 = blockIdx.y * 32 + (wave >> 2) * 16;

    const bf16_t* arow = xb + (size_t)(m0 + lh) * DD;  // A row = position m0+lh
    const bf16_t* brow = W  + (size_t)(n0 + lh) * DD;  // B col n = Wi row n (contig K)

    v8f acc = {0.f, 0.f, 0.f, 0.f, 0.f, 0.f, 0.f, 0.f};
#pragma unroll
    for (int ks = 0; ks < DD / 32; ++ks) {
        const int kb = ks * 32;
        // A 16x32 bf16: lane half ls picks K run [kb+16ls, kb+16ls+16)
        v8bf a0 = *(const v8bf*)(arow + kb + ls * 16);
        v8bf a1 = *(const v8bf*)(arow + kb + ls * 16 + 8);
        v16bf a = __builtin_shufflevector(a0, a1, 0,1,2,3,4,5,6,7,8,9,10,11,12,13,14,15);
        // B 32x16 bf16: lane half ls picks K run [kb+16ls, kb+16ls+16)
        v16bf b = *(const v16bf*)(brow + kb + ls * 16);
        acc = __builtin_amdgcn_wmma_f32_16x16x32_bf16(false, a, false, b,
                                                      (short)0, acc, false, false);
    }
    const int col = n0 + lh;
    const float bv = bias[col];
#pragma unroll
    for (int r = 0; r < 8; ++r) {                      // C: VGPR r -> M = r + 8*ls
        const int row = m0 + r + 8 * ls;
        pre[(size_t)row * NG + col] = f2bf(acc[r] + bv);
    }
}

// ------------------------------------------------------ Phase 3: LSTM scan
// grid = 64 blocks (blocks 0..31 forward, 32..63 backward), 512 thr = 16 waves.
// Block `bid` owns hidden slice [16*bid, 16*bid+16); wave (m,g) computes the
// 16x16 tile: batch rows 16m.., gate g columns. c state persists in LDS.
__global__ __launch_bounds__(512, 1) void k_lstm_scan(
    const bf16_t* __restrict__ Whf, const bf16_t* __restrict__ Whb,
    const bf16_t* __restrict__ pref, const bf16_t* __restrict__ preb,
    const float*  __restrict__ c0,  const int* __restrict__ lengths,
    bf16_t* __restrict__ hbuf, float* __restrict__ out, int* __restrict__ bar)
{
    const int dir = blockIdx.x >> 5;
    const int bid = blockIdx.x & 31;
    const int k0  = bid * 16;
    const int tid = threadIdx.x;
    const int wave = tid >> 5, lane = tid & 31;
    const int m = wave & 3, g = wave >> 2;             // batch tile / gate id
    const int lh = lane & 15, ls = lane >> 4;

    const bf16_t* Wh  = dir ? Whb  : Whf;
    const bf16_t* pre = dir ? preb : pref;
    int*    mybar = bar + dir * 32;                    // 128B apart per direction
    bf16_t* hb    = hbuf + (size_t)dir * 2 * HB;       // ping-pong pair

    __shared__ float gs[4][64][16];                    // i,f,g,o pre-activations
    __shared__ float cs[64][16];                       // cell state slice
    __shared__ int   lenS[64];

    if (tid < 64) lenS[tid] = lengths[tid];
#pragma unroll
    for (int p = 0; p < 2; ++p) {                      // c0 slice -> LDS
        int idx = tid + p * 512, b = idx >> 4, kk = idx & 15;
        cs[b][kk] = c0[(size_t)b * HH + k0 + kk];
    }

    // Preload this wave's Wh fragments for all K (held in VGPRs across steps).
    const int col = g * HH + k0 + lh;                  // gate column in [0,2048)
    const bf16_t* wrow = Wh + (size_t)col * HH;
    v16bf bw[HH / 32];
#pragma unroll
    for (int ks = 0; ks < HH / 32; ++ks)
        bw[ks] = *(const v16bf*)(wrow + ks * 32 + ls * 16);

    __syncthreads();

#pragma unroll 1
    for (int s = 0; s < TT; ++s) {
        const bf16_t* hc   = hb + (size_t)(s & 1) * HB;
        const bf16_t* arow = hc + (size_t)(16 * m + lh) * HH;

        v8f acc = {0.f, 0.f, 0.f, 0.f, 0.f, 0.f, 0.f, 0.f};
#pragma unroll
        for (int ks = 0; ks < HH / 32; ++ks) {
            v8bf a0 = *(const v8bf*)(arow + ks * 32 + ls * 16);
            v8bf a1 = *(const v8bf*)(arow + ks * 32 + ls * 16 + 8);
            v16bf a = __builtin_shufflevector(a0, a1, 0,1,2,3,4,5,6,7,8,9,10,11,12,13,14,15);
            acc = __builtin_amdgcn_wmma_f32_16x16x32_bf16(false, a, false, bw[ks],
                                                          (short)0, acc, false, false);
        }
        // add precomputed input projection (time index is per-batch for bwd)
#pragma unroll
        for (int r = 0; r < 8; ++r) {
            const int b = 16 * m + r + 8 * ls;
            const int t = dir ? ((lenS[b] + TT - 1 - s) & (TT - 1)) : s;
            gs[g][b][lh] = acc[r] + bf2f(pre[((size_t)b * TT + t) * NG + col]);
        }
        __syncthreads();

        // elementwise LSTM cell update for the 64x16 slice (2 elems/thread)
#pragma unroll
        for (int p = 0; p < 2; ++p) {
            int idx = tid + p * 512, b = idx >> 4, kk = idx & 15;
            float xi = gs[0][b][kk], xf = gs[1][b][kk];
            float xg = gs[2][b][kk], xo = gs[3][b][kk];
            float c = sigm(xf) * cs[b][kk] + sigm(xi) * tanhf(xg);
            float h = sigm(xo) * tanhf(c);
            cs[b][kk] = c;
            const int tdst = dir ? ((lenS[b] + TT - 1 - s) & (TT - 1)) : s;
            out[((size_t)b * TT + tdst) * (2 * HH) + dir * HH + k0 + kk] = h;
            hb[(size_t)((s + 1) & 1) * HB + (size_t)b * HH + k0 + kk] = f2bf(h);
        }

        // device-scope barrier across the 32 blocks of this direction
        __threadfence();
        __syncthreads();
        if (tid == 0) {
            __hip_atomic_fetch_add(mybar, 1, __ATOMIC_ACQ_REL, __HIP_MEMORY_SCOPE_AGENT);
            const int target = 32 * (s + 1);
            while (__hip_atomic_load(mybar, __ATOMIC_ACQUIRE, __HIP_MEMORY_SCOPE_AGENT) < target)
                __builtin_amdgcn_s_sleep(1);
        }
        __syncthreads();
    }
}

// ---------------------------------------------------------------------------
extern "C" void kernel_launch(void* const* d_in, const int* in_sizes, int n_in,
                              void* d_out, int out_size, void* d_ws, size_t ws_size,
                              hipStream_t stream) {
    (void)in_sizes; (void)n_in; (void)out_size; (void)ws_size;

    const float* x    = (const float*)d_in[0];
    const float* mask = (const float*)d_in[1];
    const float* h0   = (const float*)d_in[2];
    const float* c0   = (const float*)d_in[3];
    const float* Wi_f = (const float*)d_in[4];
    const float* Wh_f = (const float*)d_in[5];
    const float* b_f  = (const float*)d_in[6];
    const float* Wi_b = (const float*)d_in[7];
    const float* Wh_b = (const float*)d_in[8];
    const float* b_b  = (const float*)d_in[9];
    float* out = (float*)d_out;

    // workspace carve-up (256B aligned)
    char* ws = (char*)d_ws;
    size_t o = 0;
    int*    bar  = (int*)(ws + o);    o += 256;
    int*    len  = (int*)(ws + o);    o += 256;
    bf16_t* hbuf = (bf16_t*)(ws + o); o += (size_t)2 * 2 * HB * 2;          // 256 KB
    bf16_t* wif  = (bf16_t*)(ws + o); o += (size_t)NG * DD * 2;             // 2 MB
    bf16_t* whf  = (bf16_t*)(ws + o); o += (size_t)NG * HH * 2;
    bf16_t* wib  = (bf16_t*)(ws + o); o += (size_t)NG * DD * 2;
    bf16_t* whb  = (bf16_t*)(ws + o); o += (size_t)NG * HH * 2;
    bf16_t* xb   = (bf16_t*)(ws + o); o += (size_t)BB * TT * DD * 2;        // 32 MB
    bf16_t* pref = (bf16_t*)(ws + o); o += (size_t)BB * TT * NG * 2;        // 128 MB
    bf16_t* preb = (bf16_t*)(ws + o); o += (size_t)BB * TT * NG * 2;        // 128 MB

    // phase 0: per-call init (ws is poisoned; counters must start at 0)
    k_init_bar<<<1, 64, 0, stream>>>(bar);
    k_row_len<<<BB, 256, 0, stream>>>(mask, len);
    k_init_h<<<(2 * HB) / 512, 512, 0, stream>>>(h0, hbuf);

    // phase 1: fp32 -> bf16
    k_cvt_bf16<<<4096, 256, 0, stream>>>(x, xb, BB * TT * DD);
    k_cvt_bf16<<<1024, 256, 0, stream>>>(Wi_f, wif, NG * DD);
    k_cvt_bf16<<<1024, 256, 0, stream>>>(Wh_f, whf, NG * HH);
    k_cvt_bf16<<<1024, 256, 0, stream>>>(Wi_b, wib, NG * DD);
    k_cvt_bf16<<<1024, 256, 0, stream>>>(Wh_b, whb, NG * HH);

    // phase 2: input-projection GEMM for both directions
    dim3 g2((BB * TT) / 64, NG / 32, 2);
    k_pre_gemm<<<g2, 256, 0, stream>>>(xb, wif, wib, b_f, b_b, pref, preb);

    // phase 3: persistent bidirectional scan (fwd blocks 0-31, bwd 32-63)
    k_lstm_scan<<<64, 512, 0, stream>>>(whf, whb, pref, preb, c0, len,
                                        hbuf, out, bar);
}
